// Model_86552180949039
// MI455X (gfx1250) — matrix operand order repaired and checked by reading, not compiled, not run on hardware
//
#include <hip/hip_runtime.h>
#include <hip/hip_bf16.h>
#include <math.h>
#include <cstdint>

typedef __attribute__((ext_vector_type(16))) _Float16 v16h;
typedef __attribute__((ext_vector_type(8)))  float    v8f;
typedef __attribute__((ext_vector_type(4)))  unsigned uint32x4;
typedef __attribute__((ext_vector_type(8)))  int      int32x8;
typedef __attribute__((ext_vector_type(4)))  int      int32x4;

#define S_LEN 256
#define B_SZ  128
#define I_SZ  256
#define H_SZ  512
#define EPSC  1e-7f

struct P4 { const float* p[4]; };

union AV  { unsigned u[8]; uint4 q[2]; v16h h; };
union PK2 { unsigned u; _Float16 h[2]; };

// A-matrix 16x32 f16 layout (ISA 7.12.2): lane l holds row M=l%16;
// VGPR v holds packed K pair starting at:
__device__ __forceinline__ int a_kbase(int v, int lane) {
    int hi = (lane >= 16) ? 8 : 0;
    return (v < 4) ? (hi + 2 * v) : (16 + hi + 2 * (v - 4));
}

__device__ __forceinline__ float sigf(float x) { return 1.0f / (1.0f + expf(-x)); }

// ---------------------------------------------------------------------------
// 1) masks + state init
// ---------------------------------------------------------------------------
__global__ void init_kernel(const float* __restrict__ p_logit,
                            const float* __restrict__ ux,
                            const float* __restrict__ uh,
                            float* __restrict__ zx, float* __restrict__ zh,
                            float* __restrict__ hbuf, float* __restrict__ cbuf,
                            unsigned* __restrict__ counter) {
    int idx = blockIdx.x * blockDim.x + threadIdx.x;
    float p  = sigf(p_logit[0]);
    float lp = logf(p + EPSC) - logf(1.0f - p + EPSC);
    float inv1mp = 1.0f / (1.0f - p);
    if (idx < 4 * B_SZ * I_SZ) {
        float u = ux[idx];
        float s = sigf((lp + logf(u + EPSC) - logf(1.0f - u + EPSC)) * 10.0f);
        zx[idx] = (1.0f - s) * inv1mp;
    }
    if (idx < 4 * B_SZ * H_SZ) {
        float u = uh[idx];
        float s = sigf((lp + logf(u + EPSC) - logf(1.0f - u + EPSC)) * 10.0f);
        zh[idx] = (1.0f - s) * inv1mp;
    }
    if (idx < B_SZ * H_SZ) { hbuf[idx] = 0.0f; cbuf[idx] = 0.0f; }
    if (idx == 0) *counter = 0u;
}

// ---------------------------------------------------------------------------
// 2) pre-swizzle weights into WMMA B-operand layout, f16 pairs.
//    B[K][N] = W[g][N][K];  pair (K,K+1) is contiguous in the source row.
//    Wpk layout: [g][nt][kt][v][lane]       (read per-lane, stride-32)
//    Upk layout: [g][nt][kt][lane][v]       (8 contiguous u32 per lane ->
//                                            straight LDS image, 2x b128 reads)
// ---------------------------------------------------------------------------
__global__ void pack_kernel(P4 Wx, P4 Uw, P4 Wb, P4 Ub,
                            unsigned* __restrict__ Wpk, unsigned* __restrict__ Upk,
                            float* __restrict__ bxc, float* __restrict__ buc) {
    int idx = blockIdx.x * blockDim.x + threadIdx.x;
    if (idx < 4 * 32 * 8 * 8 * 32) {  // Wpk: K=256 -> 8 ktiles
        int lane = idx & 31, v = (idx >> 5) & 7, kt = (idx >> 8) & 7,
            nt = (idx >> 11) & 31, g = idx >> 16;
        int N = nt * 16 + (lane & 15);
        int K = kt * 32 + a_kbase(v, lane);
        const float* wsrc = Wx.p[g];
        PK2 pk;
        pk.h[0] = (_Float16)wsrc[N * I_SZ + K];
        pk.h[1] = (_Float16)wsrc[N * I_SZ + K + 1];
        Wpk[idx] = pk.u;
    }
    if (idx < 4 * 32 * 16 * 32 * 8) { // Upk: K=512 -> 16 ktiles
        int v = idx & 7, lane = (idx >> 3) & 31, kt = (idx >> 8) & 15,
            nt = (idx >> 12) & 31, g = idx >> 17;
        int N = nt * 16 + (lane & 15);
        int K = kt * 32 + a_kbase(v, lane);
        const float* wsrc = Uw.p[g];
        PK2 pk;
        pk.h[0] = (_Float16)wsrc[N * H_SZ + K];
        pk.h[1] = (_Float16)wsrc[N * H_SZ + K + 1];
        Upk[idx] = pk.u;
    }
    if (idx < 4 * H_SZ) {
        int g = idx >> 9, k = idx & (H_SZ - 1);
        bxc[idx] = Wb.p[g][k];
        buc[idx] = Ub.p[g][k];
    }
}

// ---------------------------------------------------------------------------
// 3) xproj[s,g,b,h] = sum_i (x[s,b,i]*zx[g,b,i]) * Wx[g,h,i] + bx[g,h]
//    One wave per 16x16 output tile; mask fused into the A-operand build.
// ---------------------------------------------------------------------------
__global__ void __launch_bounds__(256) xproj_kernel(
        const float* __restrict__ x, const float* __restrict__ zx,
        const unsigned* __restrict__ Wpk, const float* __restrict__ bxc,
        float* __restrict__ xproj) {
    const int lane = threadIdx.x & 31;
    const int wg   = blockIdx.x * 8 + (threadIdx.x >> 5);
    const int nt = wg & 31;            // H tile (32)
    const int mt = (wg >> 5) & 2047;   // S*B tile (2048)
    const int g  = wg >> 16;           // gate
    const int arow = mt * 16 + (lane & 15);
    const int ab   = arow & (B_SZ - 1);

    v8f acc = {};
#pragma unroll
    for (int kt = 0; kt < 8; ++kt) {
        AV a, b;
#pragma unroll
        for (int v = 0; v < 8; ++v) {
            int K = kt * 32 + a_kbase(v, lane);
            float2 xv = *(const float2*)(x  + arow * I_SZ + K);
            float2 zv = *(const float2*)(zx + (g * B_SZ + ab) * I_SZ + K);
            PK2 pk;
            pk.h[0] = (_Float16)(xv.x * zv.x);
            pk.h[1] = (_Float16)(xv.y * zv.y);
            a.u[v] = pk.u;
        }
        const unsigned* bp = Wpk + (((g * 32 + nt) * 8 + kt) << 8) + lane;
#pragma unroll
        for (int v = 0; v < 8; ++v) b.u[v] = bp[v * 32];
        acc = __builtin_amdgcn_wmma_f32_16x16x32_f16(
                false, a.h, false, b.h, (short)0, acc, false, false);
    }
    const int k = nt * 16 + (lane & 15);
    const float bias = bxc[g * H_SZ + k];
#pragma unroll
    for (int v = 0; v < 8; ++v) {
        int row = mt * 16 + ((lane < 16) ? v : v + 8);
        int s = row >> 7, bb = row & (B_SZ - 1);
        xproj[((s * 4 + g) * B_SZ + bb) * H_SZ + k] = acc[v] + bias;
    }
}

// ---------------------------------------------------------------------------
// 4) persistent recurrence. Block <-> nt (32 blocks), wave <-> mt (8 waves).
//    The block's entire B working set (4 gates x 16 ktiles x 1 KiB = 64 KiB)
//    is DMA'd into LDS ONCE by the Tensor Data Mover, then every one of the
//    256 serial timesteps reads B from LDS (2x ds_load_b128 per WMMA).
// ---------------------------------------------------------------------------
__device__ __forceinline__ void grid_barrier(unsigned* cnt, unsigned* phase) {
    __syncthreads();
    if (threadIdx.x == 0) {
        __threadfence();
        __hip_atomic_fetch_add(cnt, 1u, __ATOMIC_ACQ_REL, __HIP_MEMORY_SCOPE_AGENT);
        unsigned target = (*phase + 1u) * gridDim.x;
        while (__hip_atomic_load(cnt, __ATOMIC_ACQUIRE, __HIP_MEMORY_SCOPE_AGENT) < target) {
            __builtin_amdgcn_s_sleep(2);
        }
    }
    __syncthreads();
    __threadfence();  // acquire: make other blocks' stores visible
    *phase += 1u;
}

__global__ void __launch_bounds__(256) recurrence_kernel(
        const float* __restrict__ zh, const unsigned* __restrict__ Upk,
        const float* __restrict__ buc, const float* __restrict__ xproj,
        _Float16* __restrict__ hm, float* __restrict__ hbuf,
        float* __restrict__ cbuf, unsigned* counter, float* __restrict__ out) {
    __shared__ unsigned lbs[4 * 16 * 32 * 8];   // 64 KiB: [g][kt][lane][v]

    const int lane = threadIdx.x & 31;
    const int mt   = threadIdx.x >> 5;     // B tile (8 waves)
    const int nt   = blockIdx.x;           // H tile (32 blocks)
    const int tid  = blockIdx.x * blockDim.x + threadIdx.x;
    const int nthreads = gridDim.x * blockDim.x;
    const unsigned* hm32 = (const unsigned*)hm;
    const int arow = mt * 16 + (lane & 15);
    const int k    = nt * 16 + (lane & 15);
    const int ahi  = (lane >= 16) ? 4 : 0;  // u32 offset of this lane-half
    unsigned phase = 0;

    // ---- one-time: stage this block's B slice (Upk[g][nt][*]) into LDS ----
#if __has_builtin(__builtin_amdgcn_tensor_load_to_lds)
    if (threadIdx.x == 0) {
        // D#: 2D tile, 4 rows (gates) x 4096 dwords, row stride 131072 dwords
        uint64_t ga   = (uint64_t)(const void*)(Upk + nt * 4096);
        unsigned ldsa = (unsigned)(uintptr_t)(void*)lbs;
        uint32x4 g0 = { 1u,                          // count=1, user mode
                        ldsa,                        // lds_addr
                        (unsigned)ga,                // global_addr[31:0]
                        (unsigned)((ga >> 32) & 0x01FFFFFFu) | 0x80000000u }; // type=2
        int32x8 g1 = { 0x00020000,                   // data_size=4B
                       0x10000000,                   // tensor_dim0=4096 (lo16<<16)
                       0x00040000,                   // tensor_dim1=4 (lo16<<16)
                       0x10000000,                   // tile_dim0=4096 (<<16)
                       0x00000004,                   // tile_dim1=4
                       0x00020000,                   // tensor_dim0_stride=131072
                       0, 0 };
        int32x4 z4 = { 0, 0, 0, 0 };
        int32x8 z8 = { 0, 0, 0, 0, 0, 0, 0, 0 };
        __builtin_amdgcn_tensor_load_to_lds(g0, g1, z4, z4, z8, 0);
        __builtin_amdgcn_s_wait_tensorcnt(0);
    }
#else
    for (int i = threadIdx.x; i < 16384; i += blockDim.x) {
        int g = i >> 12;
        lbs[i] = Upk[g * 131072 + nt * 4096 + (i & 4095)];
    }
#endif
    __syncthreads();

    for (int s = 0; s < S_LEN; ++s) {
        // phase 1: per-gate masked hidden state, converted to f16
        for (int i = tid; i < 4 * B_SZ * H_SZ; i += nthreads) {
            hm[i] = (_Float16)(hbuf[i & (B_SZ * H_SZ - 1)] * zh[i]);
        }
        grid_barrier(counter, &phase);

        // keep next timestep's xproj slice flowing into L2
        if (s + 1 < S_LEN)
            __builtin_prefetch(xproj + (size_t)(s + 1) * 4 * B_SZ * H_SZ
                                     + (size_t)tid * 8, 0, 0);

        // phase 2: pre[g] = hm[g] @ U[g]^T  (f16 WMMA, f32 accum, B from LDS)
        v8f acc[4];
#pragma unroll
        for (int g = 0; g < 4; ++g) acc[g] = (v8f){};
#pragma unroll
        for (int g = 0; g < 4; ++g) {
            const unsigned* hrow = hm32 + (((g * B_SZ + arow) * H_SZ) >> 1);
#pragma unroll
            for (int kt = 0; kt < 16; ++kt) {
                AV a, b;
                // A: two contiguous 16B chunks of the hm row
                a.q[0] = *(const uint4*)(hrow + kt * 16 + ahi);
                a.q[1] = *(const uint4*)(hrow + kt * 16 + ahi + 8);
                // B: two 16B LDS reads, lane-contiguous layout
                const uint4* lb = (const uint4*)&lbs[((g * 16 + kt) * 32 + lane) * 8];
                b.q[0] = lb[0];
                b.q[1] = lb[1];
                acc[g] = __builtin_amdgcn_wmma_f32_16x16x32_f16(
                        false, a.h, false, b.h, (short)0, acc[g], false, false);
            }
        }

        // fused LSTM gate epilogue in the C/D register layout
        const float bi = buc[0 * H_SZ + k], bf = buc[1 * H_SZ + k];
        const float bo = buc[2 * H_SZ + k], bg = buc[3 * H_SZ + k];
#pragma unroll
        for (int v = 0; v < 8; ++v) {
            int b  = mt * 16 + ((lane < 16) ? v : v + 8);
            int xb = (s * 4) * B_SZ * H_SZ + b * H_SZ + k;
            float pi = acc[0][v] + bi + xproj[xb + 0 * B_SZ * H_SZ];
            float pf = acc[1][v] + bf + xproj[xb + 1 * B_SZ * H_SZ];
            float po = acc[2][v] + bo + xproj[xb + 2 * B_SZ * H_SZ];
            float pg = acc[3][v] + bg + xproj[xb + 3 * B_SZ * H_SZ];
            float ig = sigf(pi), fg = sigf(pf), og = sigf(po), gg = tanhf(pg);
            int off = b * H_SZ + k;
            float cn = fg * cbuf[off] + ig * gg;
            float hn = og * tanhf(cn);
            cbuf[off] = cn;
            hbuf[off] = hn;
            out[(size_t)s * B_SZ * H_SZ + off] = hn;
            if (s == S_LEN - 1) {
                out[(size_t)S_LEN * B_SZ * H_SZ + off] = hn;
                out[(size_t)S_LEN * B_SZ * H_SZ + B_SZ * H_SZ + off] = cn;
            }
        }
        grid_barrier(counter, &phase);
    }
}

// ---------------------------------------------------------------------------
extern "C" void kernel_launch(void* const* d_in, const int* in_sizes, int n_in,
                              void* d_out, int out_size, void* d_ws, size_t ws_size,
                              hipStream_t stream) {
    (void)in_sizes; (void)n_in; (void)out_size; (void)ws_size;
    const float* x       = (const float*)d_in[0];
    const float* p_logit = (const float*)d_in[1];
    P4 Wx = {{(const float*)d_in[2],  (const float*)d_in[4],
              (const float*)d_in[6],  (const float*)d_in[8]}};
    P4 Wb = {{(const float*)d_in[3],  (const float*)d_in[5],
              (const float*)d_in[7],  (const float*)d_in[9]}};
    P4 Uw = {{(const float*)d_in[10], (const float*)d_in[12],
              (const float*)d_in[14], (const float*)d_in[16]}};
    P4 Ub = {{(const float*)d_in[11], (const float*)d_in[13],
              (const float*)d_in[15], (const float*)d_in[17]}};
    const float* ux = (const float*)d_in[18];
    const float* uh = (const float*)d_in[19];

    char* w = (char*)d_ws;
    size_t off = 0;
    auto alloc = [&](size_t bytes) -> void* {
        off = (off + 255) & ~(size_t)255;
        void* p = w + off; off += bytes; return p;
    };
    unsigned*  counter = (unsigned*)alloc(sizeof(unsigned));
    float*     zx   = (float*)alloc(sizeof(float) * 4 * B_SZ * I_SZ);
    float*     zh   = (float*)alloc(sizeof(float) * 4 * B_SZ * H_SZ);
    unsigned*  Wpk  = (unsigned*)alloc(sizeof(unsigned) * 4 * 32 * 8 * 8 * 32);
    unsigned*  Upk  = (unsigned*)alloc(sizeof(unsigned) * 4 * 32 * 16 * 32 * 8);
    float*     bxc  = (float*)alloc(sizeof(float) * 4 * H_SZ);
    float*     buc  = (float*)alloc(sizeof(float) * 4 * H_SZ);
    _Float16*  hm   = (_Float16*)alloc(sizeof(_Float16) * 4 * B_SZ * H_SZ);
    float*     hbuf = (float*)alloc(sizeof(float) * B_SZ * H_SZ);
    float*     cbuf = (float*)alloc(sizeof(float) * B_SZ * H_SZ);
    float*     xproj= (float*)alloc(sizeof(float) * (size_t)S_LEN * 4 * B_SZ * H_SZ);

    hipLaunchKernelGGL(init_kernel, dim3(1024), dim3(256), 0, stream,
                       p_logit, ux, uh, zx, zh, hbuf, cbuf, counter);
    hipLaunchKernelGGL(pack_kernel, dim3(2048), dim3(256), 0, stream,
                       Wx, Uw, Wb, Ub, Wpk, Upk, bxc, buc);
    hipLaunchKernelGGL(xproj_kernel, dim3(32768), dim3(256), 0, stream,
                       x, zx, Wpk, bxc, xproj);
    hipLaunchKernelGGL(recurrence_kernel, dim3(32), dim3(256), 0, stream,
                       zh, Upk, buc, xproj, hm, hbuf, cbuf, counter, (float*)d_out);
}
